// MCorr1d_73821897884183
// MI455X (gfx1250) — compile-verified, matrix operand order checked by reading, not run on hardware
//
#include <hip/hip_runtime.h>
#include <hip/hip_bf16.h>

// ---- static problem dims (match reference) ----
#define X_LEN 1024
#define NB    64
#define CIN   512
#define KW    16
#define COUT  512
#define YOUT  64

// ---- tiling ----
#define COBLK   128            // C_out per workgroup (8 waves x 16)
#define ASTRIDE 40             // bf16 elems per LDS row: 32 data + 8 pad (80B, 16B aligned)
#define NCHUNK  (KW * (CIN / 32))       // 256 K-chunks of 32
#define NCOTILE (COUT / 16)             // 32 co tiles
// preconverted weight: [chunk][cotile][lane][16 bf16] = 256*32*32*16 elems
#define WFRAG_ELEMS ((size_t)NCHUNK * NCOTILE * 32 * 16)
#define WFRAG_BYTES (WFRAG_ELEMS * 2)   // 8,388,608

typedef __bf16 bf16_t;
typedef bf16_t v8bf  __attribute__((ext_vector_type(8)));
typedef bf16_t v16bf __attribute__((ext_vector_type(16)));
typedef float  v8f   __attribute__((ext_vector_type(8)));

// K index held by fragment element e for a lane-half (ISA 7.12.2, 16-bit A/B):
//   e<8 : k = 8*half + e        (VGPRs 0..3)
//   e>=8: k = 16 + 8*half + e-8 (VGPRs 4..7)
__device__ __forceinline__ int frag_k(int half, int e) {
    return (e < 8) ? (half * 8 + e) : (16 + half * 8 + (e - 8));
}

// ---- one-time (per launch) weight fp32 -> bf16 fragment re-layout ----
__global__ __launch_bounds__(256)
void preconv_weight_kernel(const float* __restrict__ wt, bf16_t* __restrict__ wf)
{
    const int gid    = blockIdx.x * 256 + (int)threadIdx.x; // 0..262143
    const int lane   = gid & 31;
    const int rest   = gid >> 5;
    const int cotile = rest & (NCOTILE - 1);
    const int chunk  = rest >> 5;          // 0..255 = w*16 + cc
    const int w      = chunk >> 4;
    const int cc     = chunk & 15;
    const int half   = lane >> 4;
    const int nlo    = lane & 15;
    const int co     = cotile * 16 + nlo;

    const float* src = wt + ((size_t)w * CIN + cc * 32) * COUT + co;
    v8bf lo8, hi8;
    #pragma unroll
    for (int e = 0; e < 8; ++e) lo8[e] = (bf16_t)src[frag_k(half, e) * COUT];
    #pragma unroll
    for (int e = 0; e < 8; ++e) hi8[e] = (bf16_t)src[frag_k(half, 8 + e) * COUT];

    bf16_t* dst = wf + (size_t)gid * 16;   // 32B per lane, contiguous
    *(v8bf*)dst       = lo8;
    *(v8bf*)(dst + 8) = hi8;
}

template <bool PRECONV>
__global__ __launch_bounds__(256)
void MCorr1d_wmma_bf16_kernel(const float* __restrict__ in,
                              const float* __restrict__ wt,
                              const float* __restrict__ bias,
                              float* __restrict__ out,
                              const bf16_t* __restrict__ wf)
{
    // double-buffered A tile: 64 rows x 32 K, bf16, padded rows
    __shared__ __align__(16) bf16_t a_lds[2][64 * ASTRIDE];

    const int y    = blockIdx.y;                 // 0..63
    const int co0w = blockIdx.x * COBLK;         // co base of workgroup
    const int tid  = (int)threadIdx.x;
    const int wave = tid >> 5;                   // 0..7 (wave32)
    const int lane = tid & 31;
    const int half = lane >> 4;
    const int nlo  = lane & 15;
    const int co   = co0w + wave * 16 + nlo;     // this lane's C_out column
    const int cotile = (co0w >> 4) + wave;

    // cooperative staging coords: 256 thr x 8 floats = 64x32 tile
    const int s_row = tid >> 2;                  // 0..63 (batch row n)
    const int s_col = (tid & 3) * 8;             // 0,8,16,24

    v8f acc[4];
    #pragma unroll
    for (int t = 0; t < 4; ++t)
        #pragma unroll
        for (int r = 0; r < 8; ++r) acc[t][r] = 0.0f;

    for (int w = 0; w < KW; ++w) {
        // mediant gather: frame index x = (y+1)(w+1)-1  (< 1024 by construction)
        const float* frame = in + (size_t)((y + 1) * (w + 1) - 1) * (NB * CIN);

        #pragma unroll 2
        for (int cc = 0; cc < CIN / 32; ++cc) {
            const int cb    = cc * 32;
            const int chunk = w * (CIN / 32) + cc;
            const int buf   = chunk & 1;

            // ---- cooperative stage: A(64x32) fp32 -> bf16 into LDS ----
            {
                const float4* g =
                    (const float4*)(frame + s_row * CIN + cb + s_col);
                float4 f0 = g[0];
                float4 f1 = g[1];
                v8bf cv;
                cv[0] = (bf16_t)f0.x; cv[1] = (bf16_t)f0.y;
                cv[2] = (bf16_t)f0.z; cv[3] = (bf16_t)f0.w;
                cv[4] = (bf16_t)f1.x; cv[5] = (bf16_t)f1.y;
                cv[6] = (bf16_t)f1.z; cv[7] = (bf16_t)f1.w;
                *(v8bf*)&a_lds[buf][s_row * ASTRIDE + s_col] = cv;
            }

            // ---- per-wave B fragment (32K x 16N), issued before barrier ----
            v16bf bfrag;
            if constexpr (PRECONV) {
                // fragment-ordered bf16: 2x contiguous 16B loads, no converts
                const bf16_t* bp =
                    wf + (((size_t)chunk * NCOTILE + cotile) * 32 + lane) * 16;
                v8bf blo = *(const v8bf*)bp;
                v8bf bhi = *(const v8bf*)(bp + 8);
                bfrag = __builtin_shufflevector(
                    blo, bhi, 0,1,2,3,4,5,6,7,8,9,10,11,12,13,14,15);
            } else {
                // strided fp32 rows of weight, convert in-register
                const float* wb = wt + ((size_t)w * CIN + cb) * COUT + co;
                #pragma unroll
                for (int e = 0; e < 16; ++e)
                    bfrag[e] = (bf16_t)wb[frag_k(half, e) * COUT];
            }

            __syncthreads();   // stores visible; barrier's DScnt wait also
                               // closes prior chunk's reads -> safe dbl-buffer

            // ---- all 4 A fragments first, then 4 WMMAs (overlap LDS lat) ----
            v16bf afrag[4];
            #pragma unroll
            for (int t = 0; t < 4; ++t) {
                const bf16_t* rp =
                    &a_lds[buf][(t * 16 + nlo) * ASTRIDE + half * 8];
                v8bf lo = *(const v8bf*)rp;          // K = 8*half .. +7
                v8bf hi = *(const v8bf*)(rp + 16);   // K = 16+8*half .. +7
                afrag[t] = __builtin_shufflevector(
                    lo, hi, 0,1,2,3,4,5,6,7,8,9,10,11,12,13,14,15);
            }
            #pragma unroll
            for (int t = 0; t < 4; ++t)
                acc[t] = __builtin_amdgcn_wmma_f32_16x16x32_bf16(
                    false, afrag[t], false, bfrag, (short)0, acc[t],
                    false, false);
        }
    }

    // ---- epilogue: bias add + store ----
    // C/D layout: VGPR r holds M = r (lanes 0-15) / M = r+8 (lanes 16-31)
    const float bv = bias[co];
    #pragma unroll
    for (int t = 0; t < 4; ++t) {
        #pragma unroll
        for (int r = 0; r < 8; ++r) {
            const int n_row = t * 16 + half * 8 + r;
            out[((size_t)y * NB + n_row) * COUT + co] = acc[t][r] + bv;
        }
    }
}

extern "C" void kernel_launch(void* const* d_in, const int* in_sizes, int n_in,
                              void* d_out, int out_size, void* d_ws, size_t ws_size,
                              hipStream_t stream) {
    (void)in_sizes; (void)n_in; (void)out_size;
    const float* in   = (const float*)d_in[0];   // [1024, 64, 512] f32
    const float* wt   = (const float*)d_in[1];   // [16, 512, 512]  f32
    const float* bias = (const float*)d_in[2];   // [512]           f32
    float* out = (float*)d_out;                  // [64, 64, 512]   f32

    dim3 grid(COUT / COBLK, YOUT);               // (4, 64) = 256 blocks
    dim3 block(256);                             // 8 wave32 waves / block

    if (ws_size >= WFRAG_BYTES && d_ws != nullptr) {
        bf16_t* wf = (bf16_t*)d_ws;
        hipLaunchKernelGGL(preconv_weight_kernel,
                           dim3((NCHUNK * NCOTILE * 32) / 256), dim3(256), 0,
                           stream, wt, wf);
        hipLaunchKernelGGL((MCorr1d_wmma_bf16_kernel<true>), grid, block, 0,
                           stream, in, wt, bias, out, (const bf16_t*)wf);
    } else {
        hipLaunchKernelGGL((MCorr1d_wmma_bf16_kernel<false>), grid, block, 0,
                           stream, in, wt, bias, out, (const bf16_t*)nullptr);
    }
}
